// MultiHeadedAttention_65824668778892
// MI455X (gfx1250) — matrix operand order repaired and checked by reading, not compiled, not run on hardware
//
#include <hip/hip_runtime.h>

// ---------------------------------------------------------------------------
// MI455X (gfx1250) implementation of dynamic-conv multi-head attention.
// All GEMM-shaped work lowered to v_wmma_f32_16x16x32_f16 (wave32 WMMA).
// f16 tile staging uses CDNA5 async global->LDS copies (ASYNCcnt), fenced
// with s_wait_asynccnt before the workgroup barrier.
// ---------------------------------------------------------------------------

typedef __attribute__((ext_vector_type(16))) _Float16 v16h;
typedef __attribute__((ext_vector_type(8)))  _Float16 v8h;
typedef __attribute__((ext_vector_type(4)))  _Float16 v4h;
typedef __attribute__((ext_vector_type(8)))  float    v8f;

#define WMMA_F16(a, b, c) \
  __builtin_amdgcn_wmma_f32_16x16x32_f16(false, (a), false, (b), (short)0, (c), false, false)

constexpr int B_ = 8, L_ = 1024, D_ = 512, H_ = 8, DH_ = 64;

// ---- CDNA5 async global->LDS copy (16B per lane, ASYNCcnt tracked) ---------
__device__ __forceinline__ void async_ld_b128(const void* g, void* lds) {
  const unsigned ldsOff = (unsigned)(unsigned long long)lds;  // wave-relative
  asm volatile("global_load_async_to_lds_b128 %0, %1, off"
               :: "v"(ldsOff), "v"(g) : "memory");
}
__device__ __forceinline__ void wait_async0() {
  asm volatile("s_wait_asynccnt 0x0" ::: "memory");
}

// ---- WMMA fragment loaders (contiguous k order, same in A and B, wave32) ----------

// A-matrix 16x32 f16: lane holds row M = lane&15. K = 16*(lane>=16) + j,
// j = 0..15 (contiguous), matching ldBT.
__device__ __forceinline__ v16h ldA(const _Float16* rowbase, int lane) {
  const int khi = (lane & 16) >> 1;  // 0 or 8
  v8h lo = *(const v8h*)(rowbase + 2 * khi);
  v8h hi = *(const v8h*)(rowbase + 2 * khi + 8);
  v16h a;
#pragma unroll
  for (int j = 0; j < 8; ++j) { a[j] = lo[j]; a[j + 8] = hi[j]; }
  return a;
}

// B-matrix 32x16 f16: lane holds col N = lane&15; K = (lane&16) + j, j = 0..15.
// Caller passes p = &T[n][c0 + (lane&16)] of a [N][K] layout. 16B aligned.
__device__ __forceinline__ v16h ldBT(const _Float16* p) {
  v8h lo = *(const v8h*)p;
  v8h hi = *(const v8h*)(p + 8);
  v16h b;
#pragma unroll
  for (int j = 0; j < 8; ++j) { b[j] = lo[j]; b[j + 8] = hi[j]; }
  return b;
}

// ---------------------------------------------------------------------------
// Kernel 0a: tiled transpose (+f32->f16 convert). out[c][r] = (f16)in[r][c].
// grid=(C/64, R/64, planes), block=128.
// ---------------------------------------------------------------------------
template <typename TIN>
__global__ __launch_bounds__(128)
void transpose_to_f16(const TIN* __restrict__ in, _Float16* __restrict__ out,
                      int R, int C, size_t inPlane, size_t outPlane) {
  __shared__ _Float16 T[64 * 72];
  const int tid = threadIdx.x;
  const int r0 = blockIdx.y * 64, c0 = blockIdx.x * 64;
  const TIN* inp = in + inPlane * blockIdx.z;
  _Float16* outp = out + outPlane * blockIdx.z;
  if constexpr (sizeof(TIN) == 4) {
    float4 xv[8];
#pragma unroll
    for (int i = 0; i < 8; ++i) {
      int idx = tid + i * 128, r = idx >> 4, c4 = (idx & 15) * 4;
      xv[i] = *(const float4*)&inp[(size_t)(r0 + r) * C + c0 + c4];
    }
#pragma unroll
    for (int i = 0; i < 8; ++i) {
      int idx = tid + i * 128, r = idx >> 4, c4 = (idx & 15) * 4;
      v4h o = {(_Float16)xv[i].x, (_Float16)xv[i].y,
               (_Float16)xv[i].z, (_Float16)xv[i].w};
      *(v4h*)&T[r * 72 + c4] = o;
    }
  } else {
#pragma unroll
    for (int i = 0; i < 4; ++i) {
      int idx = tid + i * 128, r = idx >> 3, c8 = (idx & 7) * 8;
      async_ld_b128(&inp[(size_t)(r0 + r) * C + c0 + c8], &T[r * 72 + c8]);
    }
    wait_async0();
  }
  __syncthreads();
#pragma unroll
  for (int i = 0; i < 4; ++i) {
    int idx = tid + i * 128, cc = idx >> 3, r8 = (idx & 7) * 8;
    v8h o;
#pragma unroll
    for (int j = 0; j < 8; ++j) o[j] = T[(r8 + j) * 72 + cc];
    *(v8h*)&outp[(size_t)(c0 + cc) * R + r0 + r8] = o;
  }
}

// ---------------------------------------------------------------------------
// Kernel 0b: permute conv weights to t-major [n][k] f16 (+ permuted bias).
// n = t*64 + c  maps from source column c*kw + t.
// ---------------------------------------------------------------------------
__global__ void permute_conv_w(const float* __restrict__ qkW,
                               const float* __restrict__ qkb,
                               _Float16* __restrict__ Wp, float* __restrict__ bcp,
                               int kw) {
  const int NW = kw * 64;
  for (int idx = threadIdx.x + blockIdx.x * blockDim.x; idx < NW * 64;
       idx += blockDim.x * gridDim.x) {
    int n = idx >> 6, d = idx & 63;
    int t = n >> 6, c = n & 63;
    Wp[idx] = (_Float16)qkW[d * NW + c * kw + t];
  }
  for (int n = threadIdx.x + blockIdx.x * blockDim.x; n < NW;
       n += blockDim.x * gridDim.x)
    bcp[n] = qkb[(n & 63) * kw + (n >> 6)];
}

// ---------------------------------------------------------------------------
// Kernel 1: projection GEMM  Y = (X @ W + bias) * scale  (fp32 X, f16 W^T)
// B fragments read directly from pre-transposed global WT[n][k] (L2-hot).
// Output in per-head layout ws[b][h][l][dh]. grid=(N/64, M/32), block=128.
// ---------------------------------------------------------------------------
__global__ __launch_bounds__(128)
void proj_gemm_f32(const float* __restrict__ X, const _Float16* __restrict__ WT,
                   const float* __restrict__ bias, _Float16* __restrict__ out,
                   float scale) {
  __shared__ _Float16 Xt[32 * 72];
  const int tid = threadIdx.x, lane = tid & 31, wave = tid >> 5;
  const int m0 = blockIdx.y * 32, n0 = blockIdx.x * 64;
  const int msub = wave >> 1, nsub = wave & 1;
  const int lm = lane & 15, chi = lane & 16, grp = chi >> 1;
  const _Float16* wrow0 = WT + (size_t)(n0 + nsub * 32 + lm) * D_;
  const _Float16* wrow1 = wrow0 + (size_t)16 * D_;
  v8f c0 = {}, c1 = {};
  for (int k0 = 0; k0 < D_; k0 += 64) {
    __syncthreads();
    float4 xv[4];
#pragma unroll
    for (int i = 0; i < 4; ++i) {
      int idx = tid + i * 128, r = idx >> 4, c4 = (idx & 15) * 4;
      xv[i] = *(const float4*)&X[(size_t)(m0 + r) * D_ + k0 + c4];
    }
#pragma unroll
    for (int i = 0; i < 4; ++i) {
      int idx = tid + i * 128, r = idx >> 4, c4 = (idx & 15) * 4;
      v4h o = {(_Float16)xv[i].x, (_Float16)xv[i].y,
               (_Float16)xv[i].z, (_Float16)xv[i].w};
      *(v4h*)&Xt[r * 72 + c4] = o;
    }
    __syncthreads();
#pragma unroll
    for (int cc = 0; cc < 64; cc += 32) {
      v16h a  = ldA(&Xt[(msub * 16 + lm) * 72 + cc], lane);
      v16h b0 = ldBT(wrow0 + k0 + cc + chi);
      v16h b1 = ldBT(wrow1 + k0 + cc + chi);
      c0 = WMMA_F16(a, b0, c0);
      c1 = WMMA_F16(a, b1, c1);
    }
  }
#pragma unroll
  for (int r = 0; r < 8; ++r) {
    const int row = m0 + msub * 16 + r + grp;
    const int bb = row >> 10, l = row & 1023;
#pragma unroll
    for (int ns = 0; ns < 2; ++ns) {
      const int col = n0 + nsub * 32 + ns * 16 + lm;
      const float v = ((ns ? c1[r] : c0[r]) + bias[col]) * scale;
      const int h = col >> 6, dh = col & 63;
      out[(((size_t)(bb * H_ + h) * L_ + l) << 6) + dh] = (_Float16)v;
    }
  }
}

// ---------------------------------------------------------------------------
// Kernel 2: conv-head kernel GEMM  kernT = q_h @ Wp^T + bcp  (all f16 inputs)
// Wp is pre-permuted [n][k]; B fragments direct from global. K=64.
// grid=(kw, M/32), block=128.
// ---------------------------------------------------------------------------
__global__ __launch_bounds__(128)
void conv_kern_gemm(const _Float16* __restrict__ qh, const _Float16* __restrict__ Wp,
                    const float* __restrict__ bcp, _Float16* __restrict__ outbase,
                    int kw, int head) {
  __shared__ _Float16 Xt[32 * 72];
  const int tid = threadIdx.x, lane = tid & 31, wave = tid >> 5;
  const int m0 = blockIdx.y * 32, n0 = blockIdx.x * 64;
  const int msub = wave >> 1, nsub = wave & 1;
  const int lm = lane & 15, chi = lane & 16, grp = chi >> 1;
  const int NW = kw * 64;
#pragma unroll
  for (int i = 0; i < 2; ++i) {
    int idx = tid + i * 128, r = idx >> 3, c8 = (idx & 7) * 8;
    int row = m0 + r, bb = row >> 10, l = row & 1023;
    async_ld_b128(&qh[(((size_t)(bb * H_ + head) * L_ + l) << 6) + c8],
                  &Xt[r * 72 + c8]);
  }
  wait_async0();
  __syncthreads();
  const _Float16* wrow0 = Wp + (size_t)(n0 + nsub * 32 + lm) * 64;
  const _Float16* wrow1 = wrow0 + (size_t)16 * 64;
  v8f c0 = {}, c1 = {};
#pragma unroll
  for (int cc = 0; cc < 64; cc += 32) {
    v16h a  = ldA(&Xt[(msub * 16 + lm) * 72 + cc], lane);
    v16h b0 = ldBT(wrow0 + cc + chi);
    v16h b1 = ldBT(wrow1 + cc + chi);
    c0 = WMMA_F16(a, b0, c0);
    c1 = WMMA_F16(a, b1, c1);
  }
#pragma unroll
  for (int r = 0; r < 8; ++r) {
    const int row = m0 + msub * 16 + r + grp;
#pragma unroll
    for (int ns = 0; ns < 2; ++ns) {
      const int jj = nsub * 32 + ns * 16 + lm;
      outbase[(size_t)row * NW + n0 + jj] =
          (_Float16)((ns ? c1[r] : c0[r]) + bcp[n0 + jj]);
    }
  }
}

// ---------------------------------------------------------------------------
// Kernel 3: raw head-0 scores straight into the top_attn output region.
// grid=(L/64, L/64, B), block=128 (4 waves, each a 16x64 score strip).
// ---------------------------------------------------------------------------
__global__ __launch_bounds__(128)
void head0_scores(const _Float16* __restrict__ qh, const _Float16* __restrict__ kh,
                  float* __restrict__ attn) {
  __shared__ _Float16 Kt[64 * 72];
  const int tid = threadIdx.x, lane = tid & 31, wave = tid >> 5;
  const int k0 = blockIdx.x * 64, q0 = blockIdx.y * 64 + wave * 16, bb = blockIdx.z;
  const int lm = lane & 15, chi = lane & 16, grp = chi >> 1;
  const _Float16* kbase = kh + (((size_t)(bb * H_) * L_) << 6);
#pragma unroll
  for (int i = 0; i < 4; ++i) {
    int idx = tid + i * 128, r = idx >> 3, c8 = (idx & 7) * 8;
    async_ld_b128(&kbase[((size_t)(k0 + r) << 6) + c8], &Kt[r * 72 + c8]);
  }
  wait_async0();
  __syncthreads();
  const _Float16* qrow = qh + (((size_t)(bb * H_) * L_ + q0 + lm) << 6);
  v16h a0 = ldA(qrow + 0, lane);
  v16h a1 = ldA(qrow + 32, lane);
#pragma unroll
  for (int st = 0; st < 4; ++st) {
    const _Float16* kr = &Kt[(st * 16 + lm) * 72 + chi];
    v8f c = {};
    c = WMMA_F16(a0, ldBT(kr), c);
    c = WMMA_F16(a1, ldBT(kr + 32), c);
#pragma unroll
    for (int r = 0; r < 8; ++r) {
      const int q = q0 + r + grp;
      attn[((size_t)(bb * L_ + q) << 10) + k0 + st * 16 + lm] = c[r];
    }
  }
}

// ---------------------------------------------------------------------------
// Kernel 4: in-place row softmax over L=1024. grid=B*L, block=256.
// ---------------------------------------------------------------------------
__global__ __launch_bounds__(256)
void softmax_rows(float* __restrict__ attn) {
  __shared__ float red[2][8];
  const int tid = threadIdx.x;
  float* p = attn + ((size_t)blockIdx.x << 10);
  float4 v = ((const float4*)p)[tid];
  float mx = fmaxf(fmaxf(v.x, v.y), fmaxf(v.z, v.w));
#pragma unroll
  for (int off = 16; off >= 1; off >>= 1) mx = fmaxf(mx, __shfl_xor(mx, off, 32));
  if ((tid & 31) == 0) red[0][tid >> 5] = mx;
  __syncthreads();
  float m = red[0][0];
#pragma unroll
  for (int i = 1; i < 8; ++i) m = fmaxf(m, red[0][i]);
  v.x = expf(v.x - m); v.y = expf(v.y - m); v.z = expf(v.z - m); v.w = expf(v.w - m);
  float s = v.x + v.y + v.z + v.w;
#pragma unroll
  for (int off = 16; off >= 1; off >>= 1) s += __shfl_xor(s, off, 32);
  if ((tid & 31) == 0) red[1][tid >> 5] = s;
  __syncthreads();
  float tot = 0.f;
#pragma unroll
  for (int i = 0; i < 8; ++i) tot += red[1][i];
  const float inv = 1.f / tot;
  v.x *= inv; v.y *= inv; v.z *= inv; v.w *= inv;
  ((float4*)p)[tid] = v;
}

// ---------------------------------------------------------------------------
// Kernel 5: fused flash attention (score -> online softmax -> PV) for all
// heads. Conv heads: score = sum_t kernT_t @ shift(K, t-pad)^T via haloed K
// tile in LDS. Per-row conv bias is softmax-invariant and dropped.
// K/V tiles staged with async global->LDS; V comes pre-transposed
// (vT[b][h][d][l]). grid=(L/64, nheads, B), block=128 (wave = 16 q rows).
// ---------------------------------------------------------------------------
template <int KW>
__global__ __launch_bounds__(128)
void flash_attn(const _Float16* __restrict__ Asrc,  // q_h (KW==1) or kern base
                const _Float16* __restrict__ kh, const _Float16* __restrict__ vT,
                _Float16* __restrict__ ctx, int headBase) {
  constexpr int PAD = (KW - 1) / 2;
  __shared__ _Float16 Kt[72 * 72];      // rows 72 (4-halo each side), ld 72
  __shared__ _Float16 VtT[64 * 72];     // [feature][kpos]
  __shared__ _Float16 Pt[4][16 * 64];   // per-wave probability tile
  const int tid = threadIdx.x, lane = tid & 31, wave = tid >> 5;
  const int bb = blockIdx.z, head = headBase + blockIdx.y;
  const int q0 = blockIdx.x * 64 + wave * 16;
  const int lm = lane & 15, chi = lane & 16, grp = chi >> 1;

  // invariant A fragments (Q row or dynamic-conv kernel row)
  const _Float16* arow;
  if constexpr (KW == 1) {
    arow = Asrc + (((size_t)(bb * H_ + head) * L_ + q0 + lm) << 6);
  } else {
    const int cumkw[4] = {0, 3, 6, 11};
    const int ci = head - 4;
    arow = Asrc + (size_t)cumkw[ci] * ((size_t)B_ * L_ * 64) +
           ((size_t)bb * L_ + q0 + lm) * (size_t)(KW * 64);
  }
  v16h afr[2 * KW];
#pragma unroll
  for (int t = 0; t < KW; ++t) {
    afr[2 * t]     = ldA(arow + t * 64 + 0, lane);
    afr[2 * t + 1] = ldA(arow + t * 64 + 32, lane);
  }

  const _Float16* kbase  = kh + (((size_t)(bb * H_ + head) * L_) << 6);
  const _Float16* vTbase = vT + ((size_t)(bb * H_ + head) * DH_) * L_;

  float m[8], lsum[8];
#pragma unroll
  for (int r = 0; r < 8; ++r) { m[r] = -1e30f; lsum[r] = 0.f; }
  v8f acc[4] = {};

  for (int kb = 0; kb < L_; kb += 64) {
    __syncthreads();
    // K tile with +/-4 halo (zero pad): 72 rows x 64 = 576 x 16B
#pragma unroll
    for (int i = 0; i < 5; ++i) {
      int idx = tid + i * 128;
      if (idx < 576) {
        int r = idx >> 3, c8 = (idx & 7) * 8, kp = kb - 4 + r;
        if (kp >= 0 && kp < L_) {
          async_ld_b128(&kbase[((size_t)kp << 6) + c8], &Kt[r * 72 + c8]);
        } else {
          v8h z = {};
          *(v8h*)&Kt[r * 72 + c8] = z;
        }
      }
    }
    // V tile from pre-transposed vT: 64 features x 64 kpos = 512 x 16B
#pragma unroll
    for (int i = 0; i < 4; ++i) {
      int idx = tid + i * 128, d = idx >> 3, c8 = (idx & 7) * 8;
      async_ld_b128(&vTbase[(size_t)d * L_ + kb + c8], &VtT[d * 72 + c8]);
    }
    wait_async0();
    __syncthreads();

    // scores: 16 x 64 per wave
    v8f s[4];
#pragma unroll
    for (int st = 0; st < 4; ++st) {
      v8f c = {};
#pragma unroll
      for (int t = 0; t < KW; ++t) {
        const _Float16* kr = &Kt[(st * 16 + lm + t - PAD + 4) * 72 + chi];
        c = WMMA_F16(afr[2 * t], ldBT(kr), c);
        c = WMMA_F16(afr[2 * t + 1], ldBT(kr + 32), c);
      }
      s[st] = c;
    }

    // online softmax (row stats reduced across the 16-lane half-group)
    float mx[8];
#pragma unroll
    for (int r = 0; r < 8; ++r)
      mx[r] = fmaxf(fmaxf(s[0][r], s[1][r]), fmaxf(s[2][r], s[3][r]));
#pragma unroll
    for (int off = 1; off < 16; off <<= 1)
#pragma unroll
      for (int r = 0; r < 8; ++r) mx[r] = fmaxf(mx[r], __shfl_xor(mx[r], off, 32));

    float ps[8];
#pragma unroll
    for (int r = 0; r < 8; ++r) {
      const float mn = fmaxf(m[r], mx[r]);
      const float sc = expf(m[r] - mn);
      m[r] = mn;
      lsum[r] *= sc;
#pragma unroll
      for (int st = 0; st < 4; ++st) acc[st][r] *= sc;
      float psum = 0.f;
#pragma unroll
      for (int st = 0; st < 4; ++st) {
        const float p = expf(s[st][r] - mn);
        psum += p;
        Pt[wave][(r + grp) * 64 + st * 16 + lm] = (_Float16)p;
      }
      ps[r] = psum;
    }
#pragma unroll
    for (int off = 1; off < 16; off <<= 1)
#pragma unroll
      for (int r = 0; r < 8; ++r) ps[r] += __shfl_xor(ps[r], off, 32);
#pragma unroll
    for (int r = 0; r < 8; ++r) lsum[r] += ps[r];

    // make the wave's own LDS P-tile visible across lanes before re-reading
    asm volatile("s_wait_dscnt 0x0" ::: "memory");
    __builtin_amdgcn_wave_barrier();

    // context += P @ V
    const _Float16* Pw = &Pt[wave][0];
#pragma unroll
    for (int cc = 0; cc < 64; cc += 32) {
      v16h pa = ldA(Pw + lm * 64 + cc, lane);
#pragma unroll
      for (int st = 0; st < 4; ++st) {
        v16h bv = ldBT(&VtT[(st * 16 + lm) * 72 + cc + chi]);
        acc[st] = WMMA_F16(pa, bv, acc[st]);
      }
    }
  }

  // normalize and store context in merged-head layout [b*L + q][h*64 + d]
#pragma unroll
  for (int r = 0; r < 8; ++r) {
    const float inv = 1.f / lsum[r];
    const int q = q0 + r + grp;
#pragma unroll
    for (int st = 0; st < 4; ++st) {
      const int col = head * 64 + st * 16 + lm;
      ctx[((size_t)(bb * L_ + q) << 9) + col] = (_Float16)(acc[st][r] * inv);
    }
  }
}

// ---------------------------------------------------------------------------
// Kernel 6: output projection  out = ctx @ Wo + bo  (f16 in, fp32 out)
// B fragments direct from pre-transposed global woT[n][k].
// ---------------------------------------------------------------------------
__global__ __launch_bounds__(128)
void out_proj(const _Float16* __restrict__ Xh, const _Float16* __restrict__ WT,
              const float* __restrict__ bias, float* __restrict__ out) {
  __shared__ _Float16 Xt[32 * 72];
  const int tid = threadIdx.x, lane = tid & 31, wave = tid >> 5;
  const int m0 = blockIdx.y * 32, n0 = blockIdx.x * 64;
  const int msub = wave >> 1, nsub = wave & 1;
  const int lm = lane & 15, chi = lane & 16, grp = chi >> 1;
  const _Float16* wrow0 = WT + (size_t)(n0 + nsub * 32 + lm) * D_;
  const _Float16* wrow1 = wrow0 + (size_t)16 * D_;
  v8f c0 = {}, c1 = {};
  for (int k0 = 0; k0 < D_; k0 += 64) {
    __syncthreads();
#pragma unroll
    for (int i = 0; i < 2; ++i) {
      int idx = tid + i * 128, r = idx >> 3, c8 = (idx & 7) * 8;
      async_ld_b128(&Xh[(size_t)(m0 + r) * D_ + k0 + c8], &Xt[r * 72 + c8]);
    }
    wait_async0();
    __syncthreads();
#pragma unroll
    for (int cc = 0; cc < 64; cc += 32) {
      v16h a  = ldA(&Xt[(msub * 16 + lm) * 72 + cc], lane);
      v16h b0 = ldBT(wrow0 + k0 + cc + chi);
      v16h b1 = ldBT(wrow1 + k0 + cc + chi);
      c0 = WMMA_F16(a, b0, c0);
      c1 = WMMA_F16(a, b1, c1);
    }
  }
#pragma unroll
  for (int r = 0; r < 8; ++r) {
    const int row = m0 + msub * 16 + r + grp;
#pragma unroll
    for (int ns = 0; ns < 2; ++ns) {
      const int col = n0 + nsub * 32 + ns * 16 + lm;
      out[(size_t)row * D_ + col] = (ns ? c1[r] : c0[r]) + bias[col];
    }
  }
}

// ---------------------------------------------------------------------------
extern "C" void kernel_launch(void* const* d_in, const int* in_sizes, int n_in,
                              void* d_out, int out_size, void* d_ws, size_t ws_size,
                              hipStream_t stream) {
  (void)in_sizes; (void)n_in; (void)out_size; (void)ws_size;
  const float* key   = (const float*)d_in[0];
  const float* value = (const float*)d_in[1];
  const float* query = (const float*)d_in[2];
  const float* Wk = (const float*)d_in[3];
  const float* bk = (const float*)d_in[4];
  const float* Wv = (const float*)d_in[5];
  const float* bv = (const float*)d_in[6];
  const float* Wq = (const float*)d_in[7];
  const float* bq = (const float*)d_in[8];
  const float* Wo = (const float*)d_in[9];
  const float* bo = (const float*)d_in[10];
  // conv_params flattened: h4..h7 x {qk_W, qk_b, qb_W, qb_b, bias_b}
  // (qb_W/qb_b/bias_b are per-row constants -> softmax-invariant -> unused)
  const float* qkW[4]; const float* qkb[4];
  for (int i = 0; i < 4; ++i) {
    qkW[i] = (const float*)d_in[11 + 5 * i + 0];
    qkb[i] = (const float*)d_in[11 + 5 * i + 1];
  }

  // workspace layout (f16 elems):
  //  q_h | k_h | v_h | vT | ctx | kernT(18*B*L*64) | wkT,wvT,wqT,woT | Wp | bcp
  _Float16* ws = (_Float16*)d_ws;
  const size_t QKV   = (size_t)B_ * H_ * L_ * DH_;  // 4,194,304
  const size_t KERNH = 18 * (size_t)B_ * L_ * 64;   // 9,437,184
  const size_t WSZ   = (size_t)D_ * D_;             // 262,144
  _Float16* q_h  = ws;
  _Float16* k_h  = ws + QKV;
  _Float16* v_h  = ws + 2 * QKV;
  _Float16* vT   = ws + 3 * QKV;
  _Float16* ctx  = ws + 4 * QKV;
  _Float16* kern = ws + 5 * QKV;
  _Float16* wkT  = ws + 5 * QKV + KERNH;
  _Float16* wvT  = wkT + WSZ;
  _Float16* wqT  = wvT + WSZ;
  _Float16* woT  = wqT + WSZ;
  _Float16* Wp   = woT + WSZ;                       // 1152*64 = 73,728
  float*    bcp  = (float*)(Wp + 1152 * 64);        // 1152 floats

  float* outO = (float*)d_out;                 // (B, L, D)
  float* attn = outO + (size_t)B_ * L_ * D_;   // (B, L, L) head-0 probs

  const dim3 blk(128);

  // pre-transpose weights to f16 [N][K]
  transpose_to_f16<float><<<dim3(8, 8, 1), blk, 0, stream>>>(Wk, wkT, D_, D_, 0, 0);
  transpose_to_f16<float><<<dim3(8, 8, 1), blk, 0, stream>>>(Wv, wvT, D_, D_, 0, 0);
  transpose_to_f16<float><<<dim3(8, 8, 1), blk, 0, stream>>>(Wq, wqT, D_, D_, 0, 0);
  transpose_to_f16<float><<<dim3(8, 8, 1), blk, 0, stream>>>(Wo, woT, D_, D_, 0, 0);

  const int kws[4] = {3, 3, 5, 7};
  const int cum[4] = {0, 3, 6, 11};
  const int cumn[4] = {0, 192, 384, 704};  // cumulative kw*64
  for (int ci = 0; ci < 4; ++ci) {
    permute_conv_w<<<dim3(64), dim3(256), 0, stream>>>(
        qkW[ci], qkb[ci], Wp + (size_t)cumn[ci] * 64, bcp + cumn[ci], kws[ci]);
  }

  proj_gemm_f32<<<dim3(8, 256), blk, 0, stream>>>(key,   wkT, bk, k_h, 1.0f);
  proj_gemm_f32<<<dim3(8, 256), blk, 0, stream>>>(value, wvT, bv, v_h, 1.0f);
  proj_gemm_f32<<<dim3(8, 256), blk, 0, stream>>>(query, wqT, bq, q_h, 0.125f);

  // pre-transpose V per (b,h) plane: [l][d] -> [d][l]
  transpose_to_f16<_Float16><<<dim3(1, 16, 64), blk, 0, stream>>>(
      v_h, vT, L_, DH_, (size_t)L_ * DH_, (size_t)L_ * DH_);

  for (int ci = 0; ci < 4; ++ci) {
    conv_kern_gemm<<<dim3(kws[ci], 256), blk, 0, stream>>>(
        q_h, Wp + (size_t)cumn[ci] * 64, bcp + cumn[ci],
        kern + (size_t)cum[ci] * ((size_t)B_ * L_ * 64), kws[ci], 4 + ci);
  }

  head0_scores<<<dim3(16, 16, 8), blk, 0, stream>>>(q_h, k_h, attn);
  softmax_rows<<<dim3(B_ * L_), dim3(256), 0, stream>>>(attn);

  flash_attn<1><<<dim3(16, 4, 8), blk, 0, stream>>>(q_h,  k_h, vT, ctx, 0);
  flash_attn<3><<<dim3(16, 2, 8), blk, 0, stream>>>(kern, k_h, vT, ctx, 4);
  flash_attn<5><<<dim3(16, 1, 8), blk, 0, stream>>>(kern, k_h, vT, ctx, 6);
  flash_attn<7><<<dim3(16, 1, 8), blk, 0, stream>>>(kern, k_h, vT, ctx, 7);

  out_proj<<<dim3(8, 256), blk, 0, stream>>>(ctx, woT, bo, outO);
}
